// ActionAttentionLSTM_43714177138818
// MI455X (gfx1250) — matrix operand order repaired and checked by reading, not compiled, not run on hardware
//
#include <hip/hip_runtime.h>

// ---------------------------------------------------------------------------
// ActionAttentionLSTM for MI455X (gfx1250): all convs as implicit-GEMM WMMA
// f16 x f16 -> f32 accumulate via v_wmma_f32_16x16x32_f16, wave32.
// Each wave computes a 16(M) x 32(N) output tile (template HASN2 selects the
// 16-wide variant for the 1-channel mask head). K-loop is software-pipelined:
// next fragments load while current WMMAs execute. OOB conv taps stream from
// a zeroed halo page => branchless inner loop, EXEC all-1s at every WMMA.
// ---------------------------------------------------------------------------

typedef __attribute__((ext_vector_type(16))) _Float16 v16h;
typedef __attribute__((ext_vector_type(8)))  float    v8f;
typedef __attribute__((ext_vector_type(4)))  unsigned int v4u;

union Frag16 { v16h h; v4u u[2]; };

#define T_SEG 22
#define BB    2
#define PIX   49                   // 7x7
#define M_MASK (BB*T_SEG*PIX)      // 2156 pixels (mask tower)
#define M_ST   (BB*PIX)            // 98 pixels (h0/c0/lstm)
#define ZPAGE_HALVES 4096          // >= maxCin(2560)+32

// ---------------------------------------------------------------------------
// Generic 3x3 SAME conv on 7x7 NHWC f16 activations, implicit GEMM.
//   in  : (M, Cin) f16, M enumerates (image, y, x) row-major
//   wr  : (CoutPad, 9*Cin) f16, k = tap*Cin + cin
//   zp  : zero halo page (>= Cin+32 halves of zeros)
//   MODE 0: BN(gamma,beta,mean,var rows of prm)+ReLU   MODE 1: +bias   MODE 2: raw
// ---------------------------------------------------------------------------
template<int MODE, bool HASN2>
__global__ __launch_bounds__(128) void conv3x3_wmma(
    const _Float16* __restrict__ in, const _Float16* __restrict__ wr,
    const _Float16* __restrict__ zp, const float* __restrict__ prm,
    _Float16* __restrict__ out16, float* __restrict__ out32,
    int M, int Cin, int CoutActual, int CoutPad)
{
  const int lane  = threadIdx.x;        // 0..31
  const int mlo   = lane & 15;
  const int khalf = lane >> 4;          // 0: lanes 0-15, 1: lanes 16-31
  const int tileM = blockIdx.x;
  const int tileN = blockIdx.y * blockDim.y + threadIdx.y;   // 32-wide N tile
  if (tileN * 32 >= CoutPad) return;    // uniform per wave (blockDim.x==32)

  // A-row this lane feeds (M = lane%16 of the tile)
  const int m = tileM * 16 + mlo;
  const bool mvalid = (m < M);
  int n = 0, oy = 0, ox = 0;
  if (mvalid) { n = m / PIX; int p = m - n * PIX; oy = p / 7; ox = p - oy * 7; }

  // B-columns this lane feeds (N = lane%16 of each subtile)
  const size_t kdim = (size_t)(9 * Cin);
  const _Float16* wrow0 = wr + (size_t)(tileN * 32 + mlo) * kdim + khalf * 16;
  const _Float16* wrow1 = wrow0 + 16 * kdim;   // used only when HASN2

  v8f acc0, acc1;
#pragma unroll
  for (int i = 0; i < 8; ++i) { acc0[i] = 0.0f; acc1[i] = 0.0f; }

  for (int tap = 0; tap < 9; ++tap) {
    const int ty = tap / 3, tx = tap - ty * 3;
    const int iy = oy + ty - 1, ix = ox + tx - 1;
    const bool inb = mvalid && (iy >= 0) && (iy < 7) && (ix >= 0) && (ix < 7);
    // branchless: OOB lanes stream zeros from the halo page
    const _Float16* arow =
        (inb ? (in + (size_t)(n * PIX + iy * 7 + ix) * (size_t)Cin) : zp) + 16 * khalf;
    const _Float16* wtap0 = wrow0 + (size_t)tap * (size_t)Cin;
    const _Float16* wtap1 = wrow1 + (size_t)tap * (size_t)Cin;

    // --- software-pipelined K loop: load(i+1) overlaps wmma(i) ---
    Frag16 af, bf0, bf1;
    // A 16x32 fragment in this code's k order: lane holds row M=lane%16, K=16*khalf+0..15 contiguous
    af.u[0] = *(const v4u*)(arow);
    af.u[1] = *(const v4u*)(arow + 8);
    // B 32x16 fragment in the same k order as A: lane holds column N=lane%16, K=16*khalf+0..15 contiguous
    bf0.u[0] = *(const v4u*)(wtap0);
    bf0.u[1] = *(const v4u*)(wtap0 + 8);
    if (HASN2) {
      bf1.u[0] = *(const v4u*)(wtap1);
      bf1.u[1] = *(const v4u*)(wtap1 + 8);
    }
    for (int c0 = 0; c0 + 32 < Cin; c0 += 32) {
      const int cn = c0 + 32;
      Frag16 afn, bf0n, bf1n;
      afn.u[0] = *(const v4u*)(arow + cn);
      afn.u[1] = *(const v4u*)(arow + cn + 8);
      bf0n.u[0] = *(const v4u*)(wtap0 + cn);
      bf0n.u[1] = *(const v4u*)(wtap0 + cn + 8);
      if (HASN2) {
        bf1n.u[0] = *(const v4u*)(wtap1 + cn);
        bf1n.u[1] = *(const v4u*)(wtap1 + cn + 8);
      }
      __builtin_prefetch(wtap0 + cn + 256, 0, 1);   // global_prefetch_b8
      acc0 = __builtin_amdgcn_wmma_f32_16x16x32_f16(
          false, af.h, false, bf0.h, (short)0, acc0, false, false);
      if (HASN2)
        acc1 = __builtin_amdgcn_wmma_f32_16x16x32_f16(
            false, af.h, false, bf1.h, (short)0, acc1, false, false);
      af = afn; bf0 = bf0n;
      if (HASN2) bf1 = bf1n;
    }
    acc0 = __builtin_amdgcn_wmma_f32_16x16x32_f16(
        false, af.h, false, bf0.h, (short)0, acc0, false, false);
    if (HASN2)
      acc1 = __builtin_amdgcn_wmma_f32_16x16x32_f16(
          false, af.h, false, bf1.h, (short)0, acc1, false, false);
  }

  // C/D layout: lane holds column N=lane%16, VGPR r -> row M = r + 8*khalf
  auto epilogue = [&](const v8f& acc, int co) {
    if (co >= CoutActual) return;
    float scale = 1.0f, shift = 0.0f;
    if (MODE == 0) {
      const float g  = prm[co];
      const float b  = prm[CoutActual + co];
      const float mn = prm[2 * CoutActual + co];
      const float vr = prm[3 * CoutActual + co];
      scale = g * rsqrtf(vr + 1e-5f);
      shift = b - mn * scale;
    } else if (MODE == 1) {
      shift = prm[co];
    }
#pragma unroll
    for (int r = 0; r < 8; ++r) {
      const int mr = tileM * 16 + 8 * khalf + r;
      if (mr >= M) continue;
      float v = acc[r] * scale + shift;
      if (MODE == 0) v = fmaxf(v, 0.0f);
      if (out32) out32[(size_t)mr * (size_t)CoutActual + co] = v;
      if (out16) out16[(size_t)mr * (size_t)CoutActual + co] = (_Float16)v;
    }
  };
  epilogue(acc0, tileN * 32 + mlo);
  if (HASN2) epilogue(acc1, tileN * 32 + 16 + mlo);
}

// --------------------------- data movement / glue ---------------------------

// x (B,2048,T,7,7) f32 -> xt (B*T, 7*7, 2048) f16 NHWC
__global__ void k_pack_x(const float* __restrict__ x, _Float16* __restrict__ out)
{
  const long long N = (long long)M_MASK * 2048;
  long long i = (long long)blockIdx.x * blockDim.x + threadIdx.x;
  if (i >= N) return;
  int c   = (int)(i & 2047);
  long long r = i >> 11;           // (b*T+t)*49 + pos
  int pos = (int)(r % PIX);
  int bt  = (int)(r / PIX);
  int b = bt / T_SEG, t = bt - b * T_SEG;
  out[i] = (_Float16)x[(((long long)b * 2048 + c) * T_SEG + t) * PIX + pos];
}

// w (Cout,Cin,3,3) f32 -> (CoutPad, 9*Cin) f16, k = tap*Cin + cin; pad rows zeroed
__global__ void k_pack_w(const float* __restrict__ w, _Float16* __restrict__ out,
                         int Cout, int Cin, int CoutPad)
{
  long long N = (long long)CoutPad * 9 * Cin;
  long long i = (long long)blockIdx.x * blockDim.x + threadIdx.x;
  if (i >= N) return;
  int kdim = 9 * Cin;
  int co = (int)(i / kdim);
  int r  = (int)(i - (long long)co * kdim);
  int tap = r / Cin, ci = r - tap * Cin;
  out[i] = (co < Cout) ? (_Float16)w[((long long)co * Cin + ci) * 9 + tap]
                       : (_Float16)0.0f;
}

__global__ void k_init(float* __restrict__ acc, _Float16* __restrict__ zp,
                       float* __restrict__ d_out)
{
  int i = blockIdx.x * blockDim.x + threadIdx.x;
  if (i < BB * 512) acc[i] = 0.0f;
  if (i < ZPAGE_HALVES) zp[i] = (_Float16)0.0f;
  if (i == 0) { d_out[2402] = 0.0f; d_out[2403] = 0.0f; }  // tv_loss, contrast_loss
}

// sigmoid of mask logits -> mask buffer + d_out mask region
__global__ void k_mask(const float* __restrict__ logit, float* __restrict__ mask,
                       float* __restrict__ d_out)
{
  int m = blockIdx.x * blockDim.x + threadIdx.x;
  if (m >= M_MASK) return;
  float v = 1.0f / (1.0f + __expf(-logit[m]));
  mask[m] = v;
  d_out[202 + 44 + m] = v;
}

// mx[b][t][pos][c] = mask * x  (f32, NHWC per (b,t))
__global__ void k_mx(const float* __restrict__ x, const float* __restrict__ mask,
                     float* __restrict__ mx)
{
  const long long N = (long long)M_MASK * 2048;
  long long i = (long long)blockIdx.x * blockDim.x + threadIdx.x;
  if (i >= N) return;
  int c   = (int)(i & 2047);
  long long r = i >> 11;
  int pos = (int)(r % PIX);
  int bt  = (int)(r / PIX);
  int b = bt / T_SEG, t = bt - b * T_SEG;
  mx[i] = mask[bt * PIX + pos] *
          x[(((long long)b * 2048 + c) * T_SEG + t) * PIX + pos];
}

// mean over t -> (B, 49, 2048) f16
__global__ void k_meanx(const float* __restrict__ mx, _Float16* __restrict__ out)
{
  long long i = (long long)blockIdx.x * blockDim.x + threadIdx.x;
  if (i >= (long long)M_ST * 2048) return;
  int c   = (int)(i & 2047);
  int r   = (int)(i >> 11);            // b*49 + pos
  int pos = r % PIX, b = r / PIX;
  float s = 0.0f;
  for (int t = 0; t < T_SEG; ++t)
    s += mx[(((long long)b * T_SEG + t) * PIX + pos) * 2048 + c];
  out[i] = (_Float16)(s / (float)T_SEG);
}

// att_fea[b][t] = (1/49) * sum_{pos,c} mx * wf[c]
__global__ void k_attfea(const float* __restrict__ mx, const float* __restrict__ wf,
                         float* __restrict__ attfea)
{
  __shared__ float sred[256];
  const int bt = blockIdx.x;
  float s = 0.0f;
  const float* base = mx + (long long)bt * PIX * 2048;
  for (int i = threadIdx.x; i < PIX * 2048; i += 256)
    s += base[i] * wf[i & 2047];
  sred[threadIdx.x] = s; __syncthreads();
  for (int st = 128; st > 0; st >>= 1) {
    if (threadIdx.x < st) sred[threadIdx.x] += sred[threadIdx.x + st];
    __syncthreads();
  }
  if (threadIdx.x == 0) attfea[bt] = sred[0] / (float)PIX;
}

// per-step: att_h, softmax over T -> aw (also mirrored into d_out aw region)
__global__ void k_step_att(const float* __restrict__ hbuf, const float* __restrict__ wh,
                           const float* __restrict__ attfea, float* __restrict__ aw,
                           float* __restrict__ d_out)
{
  __shared__ float sred[256];
  const int b = blockIdx.x;
  float s = 0.0f;
  const float* base = hbuf + (long long)b * PIX * 512;
  for (int i = threadIdx.x; i < PIX * 512; i += 256)
    s += base[i] * wh[i & 511];
  sred[threadIdx.x] = s; __syncthreads();
  for (int st = 128; st > 0; st >>= 1) {
    if (threadIdx.x < st) sred[threadIdx.x] += sred[threadIdx.x + st];
    __syncthreads();
  }
  if (threadIdx.x == 0) {
    float atth = sred[0] / (float)PIX;
    float l[T_SEG], mxv = -1e30f;
    for (int t = 0; t < T_SEG; ++t) { l[t] = attfea[b * T_SEG + t] + atth; mxv = fmaxf(mxv, l[t]); }
    float sum = 0.0f;
    for (int t = 0; t < T_SEG; ++t) { l[t] = __expf(l[t] - mxv); sum += l[t]; }
    for (int t = 0; t < T_SEG; ++t) {
      float a = l[t] / sum;
      aw[b * T_SEG + t] = a;
      d_out[202 + b * T_SEG + t] = a;   // last step's write survives = aws[-1]
    }
  }
}

// zin (B,49,2560) f16: ch 0..2047 = sum_t aw*mx ; ch 2048..2559 = h
__global__ void k_build_zin(const float* __restrict__ mx, const float* __restrict__ aw,
                            const float* __restrict__ hbuf, _Float16* __restrict__ zin)
{
  long long i = (long long)blockIdx.x * blockDim.x + threadIdx.x;
  if (i >= (long long)M_ST * 2560) return;
  int c   = (int)(i % 2560);
  int r   = (int)(i / 2560);          // b*49 + pos
  int pos = r % PIX, b = r / PIX;
  float v;
  if (c < 2048) {
    v = 0.0f;
    for (int t = 0; t < T_SEG; ++t)
      v += aw[b * T_SEG + t] *
           mx[(((long long)b * T_SEG + t) * PIX + pos) * 2048 + c];
  } else {
    v = hbuf[(long long)r * 512 + (c - 2048)];
  }
  zin[i] = (_Float16)v;
}

// LSTM pointwise update from gates (B,49,2048) f32 (i|f|o|g of 512 each)
__global__ void k_lstm_point(const float* __restrict__ gates,
                             float* __restrict__ hbuf, float* __restrict__ cbuf)
{
  int i = blockIdx.x * blockDim.x + threadIdx.x;
  if (i >= M_ST * 512) return;
  int c = i & 511;
  int r = i >> 9;                     // b*49 + pos
  const float* g = gates + (long long)r * 2048;
  float gi = 1.0f / (1.0f + __expf(-g[c]));
  float gf = 1.0f / (1.0f + __expf(-g[512 + c]));
  float go = 1.0f / (1.0f + __expf(-g[1024 + c]));
  float gg = tanhf(g[1536 + c]);
  float c2 = gf * cbuf[i] + gi * gg;
  float h2 = go * tanhf(c2);
  cbuf[i] = c2;
  hbuf[i] = h2;
}

// accumulate spatial mean of h into out_acc (B,512)
__global__ void k_out_acc(const float* __restrict__ hbuf, float* __restrict__ acc)
{
  int i = blockIdx.x * blockDim.x + threadIdx.x;
  if (i >= BB * 512) return;
  int c = i & 511, b = i >> 9;
  float s = 0.0f;
  for (int pos = 0; pos < PIX; ++pos)
    s += hbuf[((long long)b * PIX + pos) * 512 + c];
  acc[i] += s / (float)PIX;
}

// final FC: (acc/T) @ fc_w.T + fc_b -> d_out[0..202)
__global__ void k_fc(const float* __restrict__ acc, const float* __restrict__ fw,
                     const float* __restrict__ fb, float* __restrict__ d_out)
{
  int i = blockIdx.x * blockDim.x + threadIdx.x;
  if (i >= BB * 101) return;
  int b = i / 101, j = i - b * 101;
  float s = fb[j];
  for (int c = 0; c < 512; ++c)
    s += (acc[b * 512 + c] / (float)T_SEG) * fw[j * 512 + c];
  d_out[i] = s;
}

// ---------------------------------------------------------------------------

extern "C" void kernel_launch(void* const* d_in, const int* in_sizes, int n_in,
                              void* d_out, int out_size, void* d_ws, size_t ws_size,
                              hipStream_t stream) {
  (void)in_sizes; (void)n_in; (void)out_size; (void)ws_size;

  const float* x      = (const float*)d_in[0];
  const float* wf     = (const float*)d_in[1];
  const float* wh     = (const float*)d_in[2];
  const float* fc_w   = (const float*)d_in[3];
  const float* fc_b   = (const float*)d_in[4];
  const float* h0_w1  = (const float*)d_in[5];
  const float* h0_bn1 = (const float*)d_in[6];
  const float* h0_w2  = (const float*)d_in[7];
  const float* h0_bn2 = (const float*)d_in[8];
  const float* c0_w1  = (const float*)d_in[9];
  const float* c0_bn1 = (const float*)d_in[10];
  const float* c0_w2  = (const float*)d_in[11];
  const float* c0_bn2 = (const float*)d_in[12];
  const float* mk_w1  = (const float*)d_in[13];
  const float* mk_bn1 = (const float*)d_in[14];
  const float* mk_w2  = (const float*)d_in[15];
  const float* mk_bn2 = (const float*)d_in[16];
  const float* mk_w3  = (const float*)d_in[17];
  const float* lstm_w = (const float*)d_in[18];
  const float* lstm_b = (const float*)d_in[19];
  float* out = (float*)d_out;

  // ---- workspace carving (aligned 256B) ----
  char* ws = (char*)d_ws;
  size_t off = 0;
  auto alloc = [&](size_t bytes) -> void* {
    void* p = ws + off;
    off = (off + bytes + 255) & ~(size_t)255;
    return p;
  };
  _Float16* xt16   = (_Float16*)alloc((size_t)M_MASK * 2048 * 2);
  _Float16* wmk1   = (_Float16*)alloc((size_t)1024 * 9 * 2048 * 2);
  _Float16* wmk2   = (_Float16*)alloc((size_t)512  * 9 * 1024 * 2);
  _Float16* wmk3   = (_Float16*)alloc((size_t)16   * 9 * 512  * 2);
  _Float16* wh1    = (_Float16*)alloc((size_t)1024 * 9 * 2048 * 2);
  _Float16* wh2    = (_Float16*)alloc((size_t)512  * 9 * 1024 * 2);
  _Float16* wc1    = (_Float16*)alloc((size_t)1024 * 9 * 2048 * 2);
  _Float16* wc2    = (_Float16*)alloc((size_t)512  * 9 * 1024 * 2);
  _Float16* wls    = (_Float16*)alloc((size_t)2048 * 9 * 2560 * 2);
  _Float16* t1     = (_Float16*)alloc((size_t)M_MASK * 1024 * 2);
  _Float16* t2     = (_Float16*)alloc((size_t)M_MASK * 512  * 2);
  float*    mlogit = (float*)   alloc((size_t)M_MASK * 4);
  float*    maskb  = (float*)   alloc((size_t)M_MASK * 4);
  float*    mxb    = (float*)   alloc((size_t)M_MASK * 2048 * 4);
  _Float16* meanx  = (_Float16*)alloc((size_t)M_ST * 2048 * 2);
  _Float16* htmp   = (_Float16*)alloc((size_t)M_ST * 1024 * 2);
  float*    hbuf   = (float*)   alloc((size_t)M_ST * 512 * 4);
  float*    cbuf   = (float*)   alloc((size_t)M_ST * 512 * 4);
  _Float16* zin    = (_Float16*)alloc((size_t)M_ST * 2560 * 2);
  float*    gates  = (float*)   alloc((size_t)M_ST * 2048 * 4);
  float*    attfea = (float*)   alloc((size_t)BB * T_SEG * 4);
  float*    awbuf  = (float*)   alloc((size_t)BB * T_SEG * 4);
  float*    accout = (float*)   alloc((size_t)BB * 512 * 4);
  _Float16* zpage  = (_Float16*)alloc((size_t)ZPAGE_HALVES * 2);

  auto cdivl = [](long long a, long long b) { return (int)((a + b - 1) / b); };
  const dim3 cb32(32, 4);

  // ---- init + packing ----
  k_init<<<(ZPAGE_HALVES + 255) / 256, 256, 0, stream>>>(accout, zpage, out);
  k_pack_x<<<cdivl((long long)M_MASK * 2048, 256), 256, 0, stream>>>(x, xt16);
  k_pack_w<<<cdivl((long long)1024 * 9 * 2048, 256), 256, 0, stream>>>(mk_w1, wmk1, 1024, 2048, 1024);
  k_pack_w<<<cdivl((long long)512  * 9 * 1024, 256), 256, 0, stream>>>(mk_w2, wmk2, 512, 1024, 512);
  k_pack_w<<<cdivl((long long)16   * 9 * 512 , 256), 256, 0, stream>>>(mk_w3, wmk3, 1, 512, 16);
  k_pack_w<<<cdivl((long long)1024 * 9 * 2048, 256), 256, 0, stream>>>(h0_w1, wh1, 1024, 2048, 1024);
  k_pack_w<<<cdivl((long long)512  * 9 * 1024, 256), 256, 0, stream>>>(h0_w2, wh2, 512, 1024, 512);
  k_pack_w<<<cdivl((long long)1024 * 9 * 2048, 256), 256, 0, stream>>>(c0_w1, wc1, 1024, 2048, 1024);
  k_pack_w<<<cdivl((long long)512  * 9 * 1024, 256), 256, 0, stream>>>(c0_w2, wc2, 512, 1024, 512);
  k_pack_w<<<cdivl((long long)2048 * 9 * 2560, 256), 256, 0, stream>>>(lstm_w, wls, 2048, 2560, 2048);

  // ---- mask tower (WMMA convs): 16x32 tiles ----
  conv3x3_wmma<0, true ><<<dim3(135, 8), cb32, 0, stream>>>(xt16, wmk1, zpage, mk_bn1, t1, nullptr, M_MASK, 2048, 1024, 1024);
  conv3x3_wmma<0, true ><<<dim3(135, 4), cb32, 0, stream>>>(t1,   wmk2, zpage, mk_bn2, t2, nullptr, M_MASK, 1024, 512, 512);
  conv3x3_wmma<2, false><<<dim3(135, 1), cb32, 0, stream>>>(t2,   wmk3, zpage, nullptr, nullptr, mlogit, M_MASK, 512, 1, 16);
  k_mask<<<cdivl(M_MASK, 256), 256, 0, stream>>>(mlogit, maskb, out);

  // ---- masked features, mean over T, attention logits ----
  k_mx<<<cdivl((long long)M_MASK * 2048, 256), 256, 0, stream>>>(x, maskb, mxb);
  k_meanx<<<cdivl((long long)M_ST * 2048, 256), 256, 0, stream>>>(mxb, meanx);
  k_attfea<<<BB * T_SEG, 256, 0, stream>>>(mxb, wf, attfea);

  // ---- h0 / c0 towers (WMMA convs) ----
  conv3x3_wmma<0, true><<<dim3(7, 8), cb32, 0, stream>>>(meanx, wh1, zpage, h0_bn1, htmp, nullptr, M_ST, 2048, 1024, 1024);
  conv3x3_wmma<0, true><<<dim3(7, 4), cb32, 0, stream>>>(htmp,  wh2, zpage, h0_bn2, nullptr, hbuf, M_ST, 1024, 512, 512);
  conv3x3_wmma<0, true><<<dim3(7, 8), cb32, 0, stream>>>(meanx, wc1, zpage, c0_bn1, htmp, nullptr, M_ST, 2048, 1024, 1024);
  conv3x3_wmma<0, true><<<dim3(7, 4), cb32, 0, stream>>>(htmp,  wc2, zpage, c0_bn2, nullptr, cbuf, M_ST, 1024, 512, 512);

  // ---- ConvLSTM: 22 sequential steps ----
  for (int t = 0; t < T_SEG; ++t) {
    k_step_att<<<BB, 256, 0, stream>>>(hbuf, wh, attfea, awbuf, out);
    k_build_zin<<<cdivl((long long)M_ST * 2560, 256), 256, 0, stream>>>(mxb, awbuf, hbuf, zin);
    conv3x3_wmma<1, true><<<dim3(7, 16), cb32, 0, stream>>>(zin, wls, zpage, lstm_b, nullptr, gates, M_ST, 2560, 2048, 2048);
    k_lstm_point<<<cdivl(M_ST * 512, 256), 256, 0, stream>>>(gates, hbuf, cbuf);
    k_out_acc<<<cdivl(BB * 512, 256), 256, 0, stream>>>(hbuf, accout);
  }

  // ---- final FC ----
  k_fc<<<1, 256, 0, stream>>>(accout, fc_w, fc_b, out);
}
